// Lightweight_ConvCIFAR10_KAN_Generator_33921651704116
// MI455X (gfx1250) — matrix operand order repaired and checked by reading, not compiled, not run on hardware
//
#include <hip/hip_runtime.h>
#include <hip/hip_bf16.h>
#include <math.h>

// ---------------------------------------------------------------------------
// KAN generator on MI455X (gfx1250). Each KAN conv layer = one bf16 WMMA GEMM
//   [N, 6F] x [6F, out]
// A (built on the fly in LDS): {ReLU(p), 5 quadratic B-spline bases of p}
// B (pre-packed bf16 fragments): {base_w, spl_w*scl}
// Block = 128 threads (4 waves) -> 64 rows x 64 out-channels:
//   wave w owns out-tile w and iterates 4 row-subtiles, so each LDS expansion
//   feeds 12 WMMAs/wave/chunk (expansion cost amortized 4x vs 16-col tiles).
// ---------------------------------------------------------------------------

typedef __bf16  v16bf __attribute__((ext_vector_type(16)));
typedef float   v8f   __attribute__((ext_vector_type(8)));

#define ROWP 104   // LDS row stride in bf16 elems: 96 K-values + 8 pad (16B-aligned rows)

__device__ __forceinline__ unsigned short f2bf(float f) {
  union { float f; unsigned int u; } c; c.f = f;
  unsigned int r = c.u + 0x7FFFu + ((c.u >> 16) & 1u);   // round-to-nearest-even
  return (unsigned short)(r >> 16);
}

// ReLU + order-2 B-spline bases on the efficient-kan grid (GRID=3, k=2, [-1,1])
__device__ __forceinline__ void kan_expand(float x, float e[6]) {
  const float h   = 2.0f / 3.0f;
  const float ih  = 1.5f;    // 1/h
  const float ih2 = 0.75f;   // 1/(2h)
  e[0] = fmaxf(x, 0.0f);
  float t[8];
#pragma unroll
  for (int i = 0; i < 8; ++i) t[i] = (float)(i - 2) * h - 1.0f;
  float b0[7];
#pragma unroll
  for (int i = 0; i < 7; ++i) b0[i] = (x >= t[i] && x < t[i + 1]) ? 1.0f : 0.0f;
  float b1[6];
#pragma unroll
  for (int i = 0; i < 6; ++i)
    b1[i] = (x - t[i]) * ih * b0[i] + (t[i + 2] - x) * ih * b0[i + 1];
#pragma unroll
  for (int i = 0; i < 5; ++i)
    e[i + 1] = (x - t[i]) * ih2 * b1[i] + (t[i + 3] - x) * ih2 * b1[i + 1];
}

// ---------------------------------------------------------------------------
// 1) h0 = relu(x @ w_lin.T + b_lin)   [64,100] x [2048,100]^T -> [64,2048]
// ---------------------------------------------------------------------------
__global__ void linear_relu_kernel(const float* __restrict__ x,
                                   const float* __restrict__ w,
                                   const float* __restrict__ bias,
                                   float* __restrict__ h0) {
  int idx = blockIdx.x * blockDim.x + threadIdx.x;  // 64*2048 threads
  int nb = idx >> 11;
  int o  = idx & 2047;
  float s = bias[o];
  const float* xr = x + nb * 100;
  const float* wr = w + o * 100;
#pragma unroll 4
  for (int k = 0; k < 100; ++k) s = fmaf(xr[k], wr[k], s);
  h0[idx] = fmaxf(s, 0.0f);
}

// ---------------------------------------------------------------------------
// 2) Pack {base_w, spl_w*scl} -> bf16 WMMA B fragments (zero-padded tiles).
//    Fragment (ot, kc, s) = 512 bf16: lane l holds column (l&15), K values
//    kc*96 + s*32 + (l>>4)*16 + t.   K index k -> (f = k/6, j = k%6):
//    j==0 -> base_w[co][f], j>0 -> spl_w[co][f][j-1] * scl[co][f].
// ---------------------------------------------------------------------------
__global__ void pack_weights_kernel(const float* __restrict__ base_w,
                                    const float* __restrict__ spl_w,
                                    const float* __restrict__ scl,
                                    unsigned short* __restrict__ wp,
                                    int outC, int F, int nkc, int total) {
  int idx = blockIdx.x * blockDim.x + threadIdx.x;
  if (idx >= total) return;
  int t    = idx & 15;
  int lane = (idx >> 4) & 31;
  int frag = idx >> 9;
  int s  = frag % 3;
  int kc = (frag / 3) % nkc;
  int ot = frag / (3 * nkc);
  int col = lane & 15;
  int kb  = (lane >> 4) * 16;
  int k   = kc * 96 + s * 32 + kb + t;
  int co  = ot * 16 + col;
  float v = 0.0f;
  if (co < outC) {
    int f = k / 6, j = k - f * 6;
    if (j == 0) v = base_w[co * F + f];
    else        v = spl_w[(co * F + f) * 5 + (j - 1)] * scl[co * F + f];
  }
  wp[idx] = f2bf(v);
}

// ---------------------------------------------------------------------------
// 3) Fused up2 + unfold(3x3, pad 1) + KANLinear GEMM via v_wmma_f32_16x16x32_bf16.
// ---------------------------------------------------------------------------
__global__ __launch_bounds__(128)
void kan_layer_wmma(const float* __restrict__ prev,
                    const unsigned short* __restrict__ wp,
                    float* __restrict__ out,
                    int C, int H, int prevH, int outC, int doTanh) {
  __shared__ __align__(16) unsigned short Atile[64][ROWP];
  const int F    = C * 9;
  const int nkc  = F >> 4;
  const int lane = threadIdx.x & 31;
  const int wave = threadIdx.x >> 5;
  const int lr   = threadIdx.x >> 1;        // local row 0..63 (2 threads/row)
  const int fh   = (threadIdx.x & 1) * 8;   // this thread's 8 features of the 16
  const int HW   = H * H;
  const int n    = blockIdx.x * 64 + lr;
  const int b    = n / HW;
  const int rem  = n - b * HW;
  const int y    = rem / H;
  const int x0   = rem - y * H;
  const int pHW  = prevH * prevH;
  const float* __restrict__ pb = prev + (size_t)b * C * pHW;  // batch base

  v8f acc[4];
#pragma unroll
  for (int mt = 0; mt < 4; ++mt) acc[mt] = (v8f){0.f,0.f,0.f,0.f,0.f,0.f,0.f,0.f};

  const int ot = blockIdx.y * 4 + wave;     // this wave's 16-col out tile
  const unsigned short* __restrict__ wbase = wp + (size_t)ot * nkc * 3 * 512;

  const int arl = lane & 15;                // A-fragment row within subtile
  const int kb8 = (lane >> 4) * 16;

  // incremental decomposition of f = kc*16 + fh (+ff) into (c, r9 = f%9)
  int c0 = 0, r90 = fh;                     // fh in {0, 8}, both < 9

  for (int kc = 0; kc < nkc; ++kc) {
    __syncthreads();
    // -- gather p (fused up2 + zero-pad unfold) and expand to {relu, 5 bases} --
    int cc = c0, rr = r90;
#pragma unroll
    for (int ff = 0; ff < 8; ++ff) {
      int ky = (rr * 11) >> 5;              // rr/3 for rr in [0,8]
      int kx = rr - ky * 3;
      int yy = y + ky - 1;
      int xx = x0 + kx - 1;
      float pv = 0.0f;
      if ((unsigned)yy < (unsigned)H && (unsigned)xx < (unsigned)H)
        pv = pb[cc * pHW + (yy >> 1) * prevH + (xx >> 1)];
      float e[6];
      kan_expand(pv, e);
      unsigned short* dst = &Atile[lr][(fh + ff) * 6];
#pragma unroll
      for (int j = 0; j < 6; ++j) dst[j] = f2bf(e[j]);
      ++rr; if (rr == 9) { rr = 0; ++cc; }  // advance f by 1
    }
    r90 += 7; ++c0; if (r90 >= 9) { r90 -= 9; ++c0; }  // advance f by 16
    __syncthreads();

    // prefetch next weight chunk into cache (global_prefetch_b8)
    if (kc + 1 < nkc)
      __builtin_prefetch(wbase + (((size_t)(kc + 1) * 3) << 9), 0, 1);

    // -- K = 96 per chunk: 3 B fragments, each reused across 4 row-subtiles --
#pragma unroll
    for (int s = 0; s < 3; ++s) {
      union { uint4 q[2]; v16bf v; } bfrag;
      const uint4* wq = (const uint4*)(wbase + (((size_t)kc * 3 + s) << 9) + lane * 16);
      bfrag.q[0] = wq[0];
      bfrag.q[1] = wq[1];
#pragma unroll
      for (int mt = 0; mt < 4; ++mt) {
        union { uint4 q[2]; v16bf v; } afrag;
        afrag.q[0] = *(const uint4*)&Atile[mt * 16 + arl][s * 32 + kb8];
        afrag.q[1] = *(const uint4*)&Atile[mt * 16 + arl][s * 32 + kb8 + 8];
        acc[mt] = __builtin_amdgcn_wmma_f32_16x16x32_bf16(
            false, afrag.v, false, bfrag.v, (short)0, acc[mt], false, false);
      }
    }
  }

  // -- store: lane holds col = lane&15; rows v + 8*(lane>=16) per subtile --
  const int col = lane & 15;
  const int co  = ot * 16 + col;
  if (co < outC) {
#pragma unroll
    for (int mt = 0; mt < 4; ++mt) {
#pragma unroll
      for (int v = 0; v < 8; ++v) {
        int m  = v + ((lane >> 4) << 3);
        int n2 = blockIdx.x * 64 + mt * 16 + m;
        int b2 = n2 / HW;
        int r2 = n2 - b2 * HW;
        int y2 = r2 / H;
        int x2 = r2 - y2 * H;
        float val = acc[mt][v];
        if (doTanh) val = tanhf(val);
        out[((b2 * outC + co) * H + y2) * H + x2] = val;
      }
    }
  }
}

// ---------------------------------------------------------------------------
// Host launcher. Inputs (setup_inputs order):
//   0:x 1:w_lin 2:b_lin 3..5:base_w1,spl_w1,scl1 6..8:L2 9..11:L3 12..14:L4
// Workspace layout (bytes): h0 | a1 | a2 | a3 | wpack1..4
// ---------------------------------------------------------------------------
extern "C" void kernel_launch(void* const* d_in, const int* in_sizes, int n_in,
                              void* d_out, int out_size, void* d_ws, size_t ws_size,
                              hipStream_t stream) {
  const float* x     = (const float*)d_in[0];
  const float* w_lin = (const float*)d_in[1];
  const float* b_lin = (const float*)d_in[2];
  const float* bw[4] = {(const float*)d_in[3], (const float*)d_in[6],
                        (const float*)d_in[9], (const float*)d_in[12]};
  const float* sw[4] = {(const float*)d_in[4], (const float*)d_in[7],
                        (const float*)d_in[10], (const float*)d_in[13]};
  const float* sc[4] = {(const float*)d_in[5], (const float*)d_in[8],
                        (const float*)d_in[11], (const float*)d_in[14]};

  char* ws = (char*)d_ws;
  float* h0 = (float*)(ws + 0);          // 64*2048 f32           = 524288 B
  float* a1 = (float*)(ws + 524288);     // 64*256*4*4  f32       = 1 MB
  float* a2 = (float*)(ws + 1572864);    // 64*128*8*8  f32       = 2 MB
  float* a3 = (float*)(ws + 3670016);    // 64*64*16*16 f32       = 4 MB
  unsigned short* wpk[4] = {
      (unsigned short*)(ws + 7864320),   // L1: 16*288*3*512*2 = 14155776 B
      (unsigned short*)(ws + 22020096),  // L2:  8*144*3*512*2 =  3538944 B
      (unsigned short*)(ws + 25559040),  // L3:  4* 72*3*512*2 =   884736 B
      (unsigned short*)(ws + 26443776)   // L4:  4* 36*3*512*2 =   442368 B (padded)
  };

  const int Cs[4]     = {512, 256, 128, 64};
  const int Hs[4]     = {4, 8, 16, 32};
  const int pHs[4]    = {2, 4, 8, 16};
  const int outCs[4]  = {256, 128, 64, 3};
  const int tiles[4]  = {16, 8, 4, 4};   // 16-col tiles, padded to multiple of 4

  // 1) initial linear + relu -> h0, viewed as [64, 512, 2, 2]
  linear_relu_kernel<<<512, 256, 0, stream>>>(x, w_lin, b_lin, h0);

  // 2) pack weights to bf16 WMMA fragments (zero-padded out tiles)
  for (int l = 0; l < 4; ++l) {
    int F   = Cs[l] * 9;
    int nkc = F >> 4;
    int total = tiles[l] * nkc * 3 * 512;
    pack_weights_kernel<<<(total + 255) / 256, 256, 0, stream>>>(
        bw[l], sw[l], sc[l], wpk[l], outCs[l], F, nkc, total);
  }

  // 3) four fused KAN conv layers (WMMA GEMMs), last one writes tanh to d_out
  const float* prevs[4] = {h0, a1, a2, a3};
  float* outs[4] = {a1, a2, a3, (float*)d_out};
  for (int l = 0; l < 4; ++l) {
    int N = 64 * Hs[l] * Hs[l];
    dim3 grid(N / 64, tiles[l] / 4);     // 64 rows x 64 cols per block
    kan_layer_wmma<<<grid, 128, 0, stream>>>(prevs[l], wpk[l], outs[l],
                                             Cs[l], Hs[l], pHs[l], outCs[l],
                                             (l == 3) ? 1 : 0);
  }
  (void)in_sizes; (void)n_in; (void)out_size; (void)ws_size;
}
